// SelfAttention_46703474376880
// MI455X (gfx1250) — matrix operand order repaired and checked by reading, not compiled, not run on hardware
//
#include <hip/hip_runtime.h>
#include <hip/hip_bf16.h>

// N=4, S=2048, E=512, H=8, D=64. f16 WMMA (16x16x32) + fp32 softmax/accum.
// Wq GEMM skipped: reference discards q_proj (bug-faithful).

typedef __attribute__((ext_vector_type(16))) _Float16 v16h;
typedef __attribute__((ext_vector_type(8)))  float    v8f;

#define S_LEN 2048
#define EMB   512
#define NHEAD 8
#define HDIM  64

// ---------------------------------------------------------------------------
// Projection GEMM: out = A[M,512] @ W.T  (C[m,e] = sum_j A[m,j]*W[e,j])
// A,W fp32; out f16.  layout 0: out[n][h][s][d]   (K-proj, d-contiguous)
//                     layout 1: out[n][h][d][s]   (V-proj transposed)
// One wave computes a 16(M) x 64(E) tile; block = 4 waves = 64 M-rows.
// ---------------------------------------------------------------------------
__global__ __launch_bounds__(128) void proj_gemm(
    const float* __restrict__ A, const float* __restrict__ W,
    _Float16* __restrict__ out, int layout) {
  const int lane = threadIdx.x & 31;
  const int wave = threadIdx.x >> 5;
  const int hl   = lane >> 4;    // which 16-lane half
  const int ln   = lane & 15;
  const int mbase = blockIdx.x * 64 + wave * 16;
  const int e0    = blockIdx.y * 64;

  v8f acc[4] = {};

  for (int kb = 0; kb < EMB; kb += 32) {
    // A fragment 16x32 (contiguous k = 16*hl + s: lane half selects K sub-chunk)
    const float* arow = A + (size_t)(mbase + ln) * EMB + kb;
    v16h a;
#pragma unroll
    for (int i = 0; i < 8; ++i) {
      a[i]     = (_Float16)arow[16 * hl + i];
      a[i + 8] = (_Float16)arow[16 * hl + 8 + i];
    }
#pragma unroll
    for (int t = 0; t < 4; ++t) {
      // B fragment 32x16: column e = W row e (contiguous over j)
      const float* wrow = W + (size_t)(e0 + 16 * t + ln) * EMB + kb + 16 * hl;
      v16h b;
#pragma unroll
      for (int i = 0; i < 16; ++i) b[i] = (_Float16)wrow[i];
      acc[t] = __builtin_amdgcn_wmma_f32_16x16x32_f16(
          false, a, false, b, (short)0, acc[t], false, false);
    }
  }

#pragma unroll
  for (int t = 0; t < 4; ++t) {
    const int e = e0 + 16 * t + ln;
    const int h = e >> 6;
    const int d = e & 63;
#pragma unroll
    for (int r = 0; r < 8; ++r) {
      const int m = mbase + r + 8 * hl;
      const int n = m >> 11;        // / S_LEN
      const int s = m & (S_LEN - 1);
      size_t idx = (layout == 0)
          ? ((size_t)((n * NHEAD + h) * S_LEN + s) * HDIM + d)
          : ((size_t)((n * NHEAD + h) * HDIM + d) * S_LEN + s);
      out[idx] = (_Float16)acc[t][r];
    }
  }
}

// ---------------------------------------------------------------------------
// Flash attention: raw query (bug-faithful) vs projected K / V.
// Grid (S/64, H, N); block = 4 waves; each wave owns 16 q-rows, streams all
// 2048 keys in 64-wide tiles with online softmax. P transposed via LDS.
// ---------------------------------------------------------------------------
__global__ __launch_bounds__(128) void flash_attn(
    const float* __restrict__ query, const int* __restrict__ mask,
    const _Float16* __restrict__ kp, const _Float16* __restrict__ vt,
    _Float16* __restrict__ ao) {
  __shared__ _Float16 lds_p[4][16 * 64];

  const int lane = threadIdx.x & 31;
  const int wave = threadIdx.x >> 5;
  const int hl   = lane >> 4;
  const int ln   = lane & 15;
  const int qbase = blockIdx.x * 64 + wave * 16;
  const int h = blockIdx.y;
  const int n = blockIdx.z;
  const float scale = 0.044194173824159216f;  // 1/sqrt(512)

  // Q fragments (d-chunks 0..31, 32..63) loaded once, straight from fp32 query
  v16h aq[2];
  {
    const float* qrow = query + (size_t)(n * S_LEN + qbase + ln) * EMB + h * HDIM;
#pragma unroll
    for (int c = 0; c < 2; ++c)
#pragma unroll
      for (int i = 0; i < 8; ++i) {
        aq[c][i]     = (_Float16)qrow[32 * c + 16 * hl + i];
        aq[c][i + 8] = (_Float16)qrow[32 * c + 16 * hl + 8 + i];
      }
  }

  const _Float16* Kp = kp + (size_t)(n * NHEAD + h) * S_LEN * HDIM;  // [s][d]
  const _Float16* Vt = vt + (size_t)(n * NHEAD + h) * HDIM * S_LEN;  // [d][s]
  const int* mbase_p = mask + (size_t)n * S_LEN * S_LEN;

  v8f oacc[4] = {};
  float mstat[8], lstat[8];
#pragma unroll
  for (int r = 0; r < 8; ++r) { mstat[r] = -3.0e38f; lstat[r] = 0.0f; }
  _Float16* lp = lds_p[wave];

  for (int kb = 0; kb < S_LEN; kb += 64) {
    // ---- S = Q K^T (16x64 tile) ----
    v8f sacc[4] = {};
#pragma unroll
    for (int c = 0; c < 2; ++c)
#pragma unroll
      for (int t = 0; t < 4; ++t) {
        const _Float16* krow = Kp + (size_t)(kb + 16 * t + ln) * HDIM + 32 * c + 16 * hl;
        v16h b;
#pragma unroll
        for (int i = 0; i < 16; ++i) b[i] = krow[i];
        sacc[t] = __builtin_amdgcn_wmma_f32_16x16x32_f16(
            false, aq[c], false, b, (short)0, sacc[t], false, false);
      }

    // ---- mask + scale ----
    float p[4][8];
#pragma unroll
    for (int t = 0; t < 4; ++t)
#pragma unroll
      for (int r = 0; r < 8; ++r) {
        const int q = qbase + r + 8 * hl;
        const int k = kb + 16 * t + ln;
        const int mk = mbase_p[(size_t)q * S_LEN + k];
        const float v = sacc[t][r] * scale;
        p[t][r] = (mk == 0) ? -1.0e20f : v;
      }

    // ---- online softmax (row stats live redundantly across a 16-lane half) ----
#pragma unroll
    for (int r = 0; r < 8; ++r) {
      float tm = fmaxf(fmaxf(p[0][r], p[1][r]), fmaxf(p[2][r], p[3][r]));
#pragma unroll
      for (int off = 1; off < 16; off <<= 1)
        tm = fmaxf(tm, __shfl_xor(tm, off, 32));
      const float mnew  = fmaxf(mstat[r], tm);
      const float alpha = __expf(mstat[r] - mnew);
      mstat[r] = mnew;
      lstat[r] *= alpha;
#pragma unroll
      for (int t = 0; t < 4; ++t) oacc[t][r] *= alpha;
      float rs = 0.0f;
#pragma unroll
      for (int t = 0; t < 4; ++t) { p[t][r] = __expf(p[t][r] - mnew); rs += p[t][r]; }
#pragma unroll
      for (int off = 1; off < 16; off <<= 1) rs += __shfl_xor(rs, off, 32);
      lstat[r] += rs;
    }

    // ---- P: C-layout -> A-layout via per-wave LDS staging ----
#pragma unroll
    for (int t = 0; t < 4; ++t)
#pragma unroll
      for (int r = 0; r < 8; ++r)
        lp[(r + 8 * hl) * 64 + 16 * t + ln] = (_Float16)p[t][r];
    __asm__ volatile("s_wait_dscnt 0" ::: "memory");

    // ---- O += P V ----
#pragma unroll
    for (int c = 0; c < 2; ++c) {
      v16h ap;
#pragma unroll
      for (int i = 0; i < 8; ++i) {
        ap[i]     = lp[ln * 64 + 32 * c + 16 * hl + i];
        ap[i + 8] = lp[ln * 64 + 32 * c + 16 * hl + 8 + i];
      }
#pragma unroll
      for (int t = 0; t < 4; ++t) {
        const _Float16* vrow = Vt + (size_t)(16 * t + ln) * S_LEN + kb + 32 * c + 16 * hl;
        v16h b;
#pragma unroll
        for (int i = 0; i < 16; ++i) b[i] = vrow[i];
        oacc[t] = __builtin_amdgcn_wmma_f32_16x16x32_f16(
            false, ap, false, b, (short)0, oacc[t], false, false);
      }
    }
  }

  // epilogue: normalize and scatter to [n][q][h*64+d] (f16)
#pragma unroll
  for (int t = 0; t < 4; ++t) {
    const int d = h * HDIM + 16 * t + ln;
#pragma unroll
    for (int r = 0; r < 8; ++r) {
      const int q = qbase + r + 8 * hl;
      ao[(size_t)(n * S_LEN + q) * EMB + d] = (_Float16)(oacc[t][r] / lstat[r]);
    }
  }
}

// ---------------------------------------------------------------------------
// Output projection: out = attn(f16) @ Wo.T + bo   (fp32 output)
// ---------------------------------------------------------------------------
__global__ __launch_bounds__(128) void oproj_gemm(
    const _Float16* __restrict__ A, const float* __restrict__ W,
    const float* __restrict__ bias, float* __restrict__ out) {
  const int lane = threadIdx.x & 31;
  const int wave = threadIdx.x >> 5;
  const int hl   = lane >> 4;
  const int ln   = lane & 15;
  const int mbase = blockIdx.x * 64 + wave * 16;
  const int e0    = blockIdx.y * 64;

  v8f acc[4] = {};

  for (int kb = 0; kb < EMB; kb += 32) {
    const _Float16* arow = A + (size_t)(mbase + ln) * EMB + kb;
    v16h a;
#pragma unroll
    for (int i = 0; i < 8; ++i) {
      a[i]     = arow[16 * hl + i];
      a[i + 8] = arow[16 * hl + 8 + i];
    }
#pragma unroll
    for (int t = 0; t < 4; ++t) {
      const float* wrow = W + (size_t)(e0 + 16 * t + ln) * EMB + kb + 16 * hl;
      v16h b;
#pragma unroll
      for (int i = 0; i < 16; ++i) b[i] = (_Float16)wrow[i];
      acc[t] = __builtin_amdgcn_wmma_f32_16x16x32_f16(
          false, a, false, b, (short)0, acc[t], false, false);
    }
  }

#pragma unroll
  for (int t = 0; t < 4; ++t) {
    const int e = e0 + 16 * t + ln;
    const float bv = bias[e];
#pragma unroll
    for (int r = 0; r < 8; ++r) {
      const int m = mbase + r + 8 * hl;
      out[(size_t)m * EMB + e] = acc[t][r] + bv;
    }
  }
}

// ---------------------------------------------------------------------------
extern "C" void kernel_launch(void* const* d_in, const int* in_sizes, int n_in,
                              void* d_out, int out_size, void* d_ws, size_t ws_size,
                              hipStream_t stream) {
  const float* values = (const float*)d_in[0];
  const float* keys   = (const float*)d_in[1];
  const float* query  = (const float*)d_in[2];
  const int*   mask   = (const int*)d_in[3];
  // d_in[4] = Wq: intentionally unused (reference discards q_proj)
  const float* Wk = (const float*)d_in[5];
  const float* Wv = (const float*)d_in[6];
  const float* Wo = (const float*)d_in[7];
  const float* bo = (const float*)d_in[8];
  float* out = (float*)d_out;

  char* ws = (char*)d_ws;
  _Float16* kp = (_Float16*)(ws);                         // [N,H,S,D]  8 MB
  _Float16* vt = (_Float16*)(ws + (size_t)8  * 1048576);  // [N,H,D,S]  8 MB
  _Float16* ao = (_Float16*)(ws + (size_t)16 * 1048576);  // [N,S,E]    8 MB

  dim3 gp(128, 8), bp(128);
  proj_gemm<<<gp, bp, 0, stream>>>(keys,   Wk, kp, 0);
  proj_gemm<<<gp, bp, 0, stream>>>(values, Wv, vt, 1);

  dim3 ga(S_LEN / 64, NHEAD, 4);
  flash_attn<<<ga, bp, 0, stream>>>(query, mask, kp, vt, ao);

  oproj_gemm<<<gp, bp, 0, stream>>>(ao, Wo, bo, out);
}
